// PixPro_67516885893305
// MI455X (gfx1250) — matrix operand order repaired and checked by reading, not compiled, not run on hardware
//
#include <hip/hip_runtime.h>
#include <hip/hip_bf16.h>
#include <stdint.h>

// ---------------------------------------------------------------------------
// PixPro forward on MI455X (gfx1250).
//  - All GEMMs: v_wmma_f32_16x16x32_bf16, wave32, 64x128 block tiles.
//  - All GEMM operands are bf16 in async-friendly layout (A row-major [M,K],
//    B K-major [N,K]); f32->bf16 conversion happens exactly once per element
//    in bandwidth-bound pre-passes.
//  - Main loop: 3-stage software pipeline of global_load_async_to_lds_b128
//    (ASYNCcnt) with branch-free steady state: issue tile t+2, wait for tile
//    t (s_wait_asynccnt 6 = newest two tiles in flight), barrier, WMMA.
// ---------------------------------------------------------------------------

#define BATCH 32
#define CIN   1024
#define HW    1024   // 32*32
#define D1    2048
#define D2    256
#define CB    8      // batch chunk for the encoder stages
#define NCHUNK (BATCH / CB)

#define BM 64
#define BN 128
#define BK 32
#define AKP (BK + 8)   // LDS row = 40 shorts = 80B: 16B-aligned for b128 async
#define STAGES 3

typedef __bf16 bf16_t;
typedef bf16_t v16bf __attribute__((ext_vector_type(16)));
typedef float  v8f   __attribute__((ext_vector_type(8)));

union FragAB { uint32_t u[8]; v16bf v; };
union Pack8  { unsigned short s[8]; uint4 q; };

__device__ __forceinline__ unsigned short f32_to_bf16(float f) {
    uint32_t u = __builtin_bit_cast(uint32_t, f);
    uint32_t r = u + 0x7FFFu + ((u >> 16) & 1u);   // round-to-nearest-even
    return (unsigned short)(r >> 16);
}

__device__ __forceinline__ void async_copy_b128(const void* gsrc, void* lds) {
    uint32_t loff = (uint32_t)(uintptr_t)lds;          // wave-relative LDS offset
    uint64_t ga   = (uint64_t)(uintptr_t)gsrc;
    asm volatile("global_load_async_to_lds_b128 %0, %1, off"
                 :: "v"(loff), "v"(ga) : "memory");
}
__device__ __forceinline__ void wait_async6() {
    asm volatile("s_wait_asynccnt 0x6" ::: "memory");
}
__device__ __forceinline__ void wait_async3() {
    asm volatile("s_wait_asynccnt 0x3" ::: "memory");
}
__device__ __forceinline__ void wait_async0() {
    asm volatile("s_wait_asynccnt 0x0" ::: "memory");
}

// ---------------------------------------------------------------------------
// Tiled WMMA GEMM:  C[M,N] = act(A * B + bias)
//   A : bf16 row-major [M,K] (lda), per-batch stride aBatch
//   B : bf16 K-major   [N,K] (ldb), per-batch stride bBatch
//   BIAS : 0 none, 1 bias[m], 2 bias[n]
//   ACT  : 0 none, 1 relu, 2 relu^2
//   OUT_F32   : f32 row-major store  Cf[m*ldc + n]
//   OUT_F32_T : f32 K-major store    Cf[n*ldct + m]   (2x float4 per frag)
//   OUT_BF16_T: bf16 K-major store   Ch[n*ldct + m]   (1x uint4  per frag)
// Requires M%BM==0, N%BN==0, K%BK==0, K/BK >= 2 (true for every stage here).
// ---------------------------------------------------------------------------
template <int BIAS, int ACT, bool OUT_F32, bool OUT_F32_T, bool OUT_BF16_T>
__global__ __launch_bounds__(256)
void gemm_wmma(const unsigned short* __restrict__ Ag, long long aBatch, int lda,
               const unsigned short* __restrict__ Bg, long long bBatch, int ldb,
               const float* __restrict__ bias,
               float* __restrict__ Cf, long long cfBatch, int ldc,
               unsigned short* __restrict__ Ch, long long chBatch, int ldct,
               int K)
{
    __shared__ unsigned short As[STAGES][BM][AKP];
    __shared__ unsigned short Bs[STAGES][BN][AKP];

    const int tid  = threadIdx.x;
    const int lane = tid & 31;
    const int wv   = tid >> 5;          // 8 waves
    const int wvM  = wv >> 2;           // 0..1  -> 32-row slab
    const int wvN  = wv & 3;            // 0..3  -> 32-col slab
    const int half = lane >> 4;
    const int l15  = lane & 15;

    const int n0 = blockIdx.x * BN;
    const int m0 = blockIdx.y * BM;
    const int bz = blockIdx.z;

    // per-thread async chunk assignment (one b128 for A, two for B per tile)
    const int rowA = tid >> 2, kkA = (tid & 3) * 8;     // 64 rows x 4 chunks
    const int colB = tid >> 1, kkB = (tid & 1) * 16;    // 128 rows x 2 pairs
    const unsigned short* Abase =
        Ag + (size_t)bz * aBatch + (size_t)(m0 + rowA) * lda + kkA;
    const unsigned short* Bbase =
        Bg + (size_t)bz * bBatch + (size_t)(n0 + colB) * ldb + kkB;

    auto issue = [&](int k0, int buf) {
        async_copy_b128(Abase + k0, &As[buf][rowA][kkA]);
        async_copy_b128(Bbase + k0,     &Bs[buf][colB][kkB]);
        async_copy_b128(Bbase + k0 + 8, &Bs[buf][colB][kkB + 8]);
    };

    v8f acc[2][2] = {};

    auto compute = [&](int buf) {
        FragAB a[2], b[2];
        #pragma unroll
        for (int mi = 0; mi < 2; ++mi) {
            int mrow = wvM * 32 + mi * 16 + l15;
            #pragma unroll
            for (int v = 0; v < 8; ++v) {
                int kk = (v >> 2) * 16 + half * 8 + (v & 3) * 2;
                a[mi].u[v] = *(const uint32_t*)&As[buf][mrow][kk];
            }
        }
        #pragma unroll
        for (int ni = 0; ni < 2; ++ni) {
            int ncol = wvN * 32 + ni * 16 + l15;
            #pragma unroll
            for (int v = 0; v < 8; ++v) {
                int kk = (v >> 2) * 16 + half * 8 + (v & 3) * 2;
                b[ni].u[v] = *(const uint32_t*)&Bs[buf][ncol][kk];
            }
        }
        #pragma unroll
        for (int mi = 0; mi < 2; ++mi)
            #pragma unroll
            for (int ni = 0; ni < 2; ++ni)
                acc[mi][ni] = __builtin_amdgcn_wmma_f32_16x16x32_bf16(
                    false, a[mi].v, false, b[ni].v,
                    (short)0, acc[mi][ni], false, false);
    };

    const int nIter = K / BK;   // >= 2 always (min K = 256 -> 8)
    issue(0, 0);
    issue(BK, 1);
    int cur = 0, ins = 2;
    // branch-free steady state: issue t+2, retire t, compute t
    for (int t = 0; t < nIter - 2; ++t) {
        issue((t + 2) * BK, ins);
        wait_async6();              // newest two tiles may stay in flight
        __syncthreads();
        compute(cur);
        __syncthreads();
        cur = (cur == STAGES - 1) ? 0 : cur + 1;
        ins = (ins == STAGES - 1) ? 0 : ins + 1;
    }
    // epilogue: tiles nIter-2 and nIter-1 (no further issues)
    wait_async3();
    __syncthreads();
    compute(cur);
    cur = (cur == STAGES - 1) ? 0 : cur + 1;
    wait_async0();
    __syncthreads();
    compute(cur);

    // ---------------- epilogue stores --------------------------------------
    #pragma unroll
    for (int mi = 0; mi < 2; ++mi) {
        #pragma unroll
        for (int ni = 0; ni < 2; ++ni) {
            int nCol   = n0 + wvN * 32 + ni * 16 + l15;
            int mBase8 = m0 + wvM * 32 + mi * 16 + half * 8;
            float o[8];
            Pack8 pk;
            #pragma unroll
            for (int v = 0; v < 8; ++v) {
                int mRow = mBase8 + v;
                float val = acc[mi][ni][v];
                if (BIAS == 1) val += bias[mRow];
                if (BIAS == 2) val += bias[nCol];
                if (ACT == 1) val = fmaxf(val, 0.0f);
                if (ACT == 2) { val = fmaxf(val, 0.0f); val = val * val; }
                o[v] = val;
                if (OUT_F32)
                    Cf[(size_t)bz * cfBatch + (size_t)mRow * ldc + nCol] = val;
                if (OUT_BF16_T) pk.s[v] = f32_to_bf16(val);
            }
            if (OUT_BF16_T)    // 8 consecutive m -> one 16B store
                *(uint4*)(Ch + (size_t)bz * chBatch +
                          (size_t)nCol * ldct + mBase8) = pk.q;
            if (OUT_F32_T) {   // 8 consecutive m -> two 16B stores
                float4* dst = (float4*)(Cf + (size_t)bz * cfBatch +
                                        (size_t)nCol * ldct + mBase8);
                dst[0] = make_float4(o[0], o[1], o[2], o[3]);
                dst[1] = make_float4(o[4], o[5], o[6], o[7]);
            }
        }
    }
}

// ---------------------------------------------------------------------------
// Tiled transpose + f32->bf16: X[b][c][hw] -> XT[b][hw][c]. 32x32 tiles.
// ---------------------------------------------------------------------------
__global__ __launch_bounds__(256)
void transpose_conv_kernel(const float* __restrict__ X,
                           unsigned short* __restrict__ XT)
{
    __shared__ unsigned short T[32][34];
    int bz = blockIdx.z;
    int hw0 = blockIdx.x * 32, c0 = blockIdx.y * 32;
    int tx = threadIdx.x & 31, ty = threadIdx.x >> 5;
    const float* src = X + (size_t)bz * CIN * HW;
    #pragma unroll
    for (int p = 0; p < 4; ++p) {
        int c = ty + p * 8;
        T[c][tx] = f32_to_bf16(src[(size_t)(c0 + c) * HW + hw0 + tx]);
    }
    __syncthreads();
    unsigned short* dst = XT + (size_t)bz * HW * CIN;
    #pragma unroll
    for (int p = 0; p < 4; ++p) {
        int hw = ty + p * 8;
        dst[(size_t)(hw0 + hw) * CIN + c0 + tx] = T[tx][hw];
    }
}

// ---------------------------------------------------------------------------
// Elementwise f32 -> bf16 (weights), 8 elements / thread, 16B stores.
// ---------------------------------------------------------------------------
__global__ void convw_kernel(const float* __restrict__ src,
                             unsigned short* __restrict__ dst, int n8)
{
    int i = blockIdx.x * 256 + threadIdx.x;
    if (i >= n8) return;
    const float* s = src + (size_t)i * 8;
    Pack8 pk;
    #pragma unroll
    for (int j = 0; j < 8; ++j) pk.s[j] = f32_to_bf16(s[j]);
    *(uint4*)(dst + (size_t)i * 8) = pk.q;
}

// ---------------------------------------------------------------------------
// Column L2-normalize, writing xn K-major: xnT[i][c] = x[c,i]/(||x[:,i]||+1e-8)
// ---------------------------------------------------------------------------
__global__ __launch_bounds__(256)
void colnorm_kernel(const float* __restrict__ xe, unsigned short* __restrict__ xnT)
{
    int b  = blockIdx.y;
    int wv = threadIdx.x >> 5, lane = threadIdx.x & 31;
    int i  = blockIdx.x * 8 + wv;
    const float* base = xe + (size_t)b * D2 * HW + i;
    float vals[8];
    float ss = 0.0f;
    #pragma unroll
    for (int j = 0; j < 8; ++j) {
        float v = base[(size_t)(lane + j * 32) * HW];
        vals[j] = v;
        ss += v * v;
    }
    #pragma unroll
    for (int s = 16; s > 0; s >>= 1) ss += __shfl_xor(ss, s, 32);
    float inv = 1.0f / (sqrtf(ss) + 1e-8f);
    unsigned short* ob = xnT + (size_t)b * HW * D2 + (size_t)i * D2;
    #pragma unroll
    for (int j = 0; j < 8; ++j)
        ob[lane + j * 32] = f32_to_bf16(vals[j] * inv);
}

// ---------------------------------------------------------------------------
// A matrix: A[k1,k2] = (dist < 0.7) ? 1 : 0 (stored as float).
// ---------------------------------------------------------------------------
__global__ void amat_kernel(const float* __restrict__ p1,
                            const float* __restrict__ p2,
                            float* __restrict__ out)
{
    int k2 = blockIdx.x * blockDim.x + threadIdx.x;
    int k1 = blockIdx.y * blockDim.y + threadIdx.y;
    float x1 = p1[0], y1 = p1[1], w1 = p1[2], h1 = p1[3];
    float x2 = p2[0], y2 = p2[1];
    float rw = w1 * (1.0f / 32.0f), rh = h1 * (1.0f / 32.0f);
    float diag = sqrtf(w1 * w1 + h1 * h1);
    int i1 = k1 & 31, j1 = k1 >> 5;
    int i2 = k2 & 31, j2 = k2 >> 5;
    float dx = (x1 + i1 * rw) - (x2 + i2 * rw);
    float dy = (y1 + j1 * rh) - (y2 + j2 * rh);
    float dist = sqrtf(dx * dx + dy * dy) / diag;
    out[(size_t)k1 * 1024 + k2] = (dist < 0.7f) ? 1.0f : 0.0f;
}

// ---------------------------------------------------------------------------
extern "C" void kernel_launch(void* const* d_in, const int* in_sizes, int n_in,
                              void* d_out, int out_size, void* d_ws, size_t ws_size,
                              hipStream_t stream)
{
    (void)in_sizes; (void)n_in; (void)out_size; (void)ws_size;
    typedef long long ll;

    const float* x1      = (const float*)d_in[0];
    const float* x2      = (const float*)d_in[1];
    const float* p1      = (const float*)d_in[2];
    const float* p2      = (const float*)d_in[3];
    const float* w_proj1 = (const float*)d_in[6];
    const float* b_proj1 = (const float*)d_in[7];
    const float* w_proj2 = (const float*)d_in[8];
    const float* b_proj2 = (const float*)d_in[9];
    const float* w_ppm   = (const float*)d_in[10];
    const float* b_ppm   = (const float*)d_in[11];

    float* out_x1e = (float*)d_out;
    float* out_x2e = out_x1e + (size_t)BATCH * D2 * HW;
    float* out_y   = out_x2e + (size_t)BATCH * D2 * HW;
    float* out_A   = out_y   + (size_t)BATCH * D2 * HW;

    // Workspace (peak ~117MB):
    //   region [0,96MB): encodes: XT(16MB)+H1T(32MB); PPM: xnT(16)+S(64)+g(16)
    //   [96MB): x1eT 16MB; [112MB): w1h 4MB; [116MB): w2h 1MB; [117MB): wph
    char* ws = (char*)d_ws;
    const size_t MB = 1024ull * 1024ull;
    unsigned short* XT   = (unsigned short*)(ws);
    unsigned short* H1T  = (unsigned short*)(ws + (size_t)CB * HW * CIN * 2);
    unsigned short* xnT  = (unsigned short*)(ws);
    unsigned short* Smat = (unsigned short*)(ws + (size_t)BATCH * HW * D2 * 2);
    unsigned short* g    = (unsigned short*)(ws + (size_t)BATCH * HW * D2 * 2
                                                + (size_t)BATCH * HW * HW * 2);
    unsigned short* x1eT = (unsigned short*)(ws + 96 * MB);
    unsigned short* w1h  = (unsigned short*)(ws + 112 * MB);
    unsigned short* w2h  = (unsigned short*)(ws + 116 * MB);
    unsigned short* wph  = (unsigned short*)(ws + 117 * MB);

    dim3 blk(256);

    // 0) convert weights to bf16 once
    convw_kernel<<<dim3((D1 * CIN / 8 + 255) / 256), blk, 0, stream>>>(
        w_proj1, w1h, D1 * CIN / 8);
    convw_kernel<<<dim3((D2 * D1 / 8 + 255) / 256), blk, 0, stream>>>(
        w_proj2, w2h, D2 * D1 / 8);
    convw_kernel<<<dim3((D2 * D2 / 8 + 255) / 256), blk, 0, stream>>>(
        w_ppm, wph, D2 * D2 / 8);

    // 1) encoder for x1, in batch chunks (XT/H1T reused per chunk)
    for (int c = 0; c < NCHUNK; ++c) {
        transpose_conv_kernel<<<dim3(HW / 32, CIN / 32, CB), blk, 0, stream>>>(
            x1 + (size_t)c * CB * CIN * HW, XT);
        gemm_wmma<1, 1, false, false, true>
            <<<dim3(HW / BN, D1 / BM, CB), blk, 0, stream>>>(
            w1h, 0, CIN, XT, (ll)HW * CIN, CIN, b_proj1,
            nullptr, 0, 0, H1T, (ll)HW * D1, D1, CIN);
        gemm_wmma<1, 0, true, false, true>
            <<<dim3(HW / BN, D2 / BM, CB), blk, 0, stream>>>(
            w2h, 0, D1, H1T, (ll)HW * D1, D1, b_proj2,
            out_x1e + (size_t)c * CB * D2 * HW, (ll)D2 * HW, HW,
            x1eT + (size_t)c * CB * HW * D2, (ll)HW * D2, D2, D1);
    }
    // 2) encoder for x2 (f32 out only)
    for (int c = 0; c < NCHUNK; ++c) {
        transpose_conv_kernel<<<dim3(HW / 32, CIN / 32, CB), blk, 0, stream>>>(
            x2 + (size_t)c * CB * CIN * HW, XT);
        gemm_wmma<1, 1, false, false, true>
            <<<dim3(HW / BN, D1 / BM, CB), blk, 0, stream>>>(
            w1h, 0, CIN, XT, (ll)HW * CIN, CIN, b_proj1,
            nullptr, 0, 0, H1T, (ll)HW * D1, D1, CIN);
        gemm_wmma<1, 0, true, false, false>
            <<<dim3(HW / BN, D2 / BM, CB), blk, 0, stream>>>(
            w2h, 0, D1, H1T, (ll)HW * D1, D1, b_proj2,
            out_x2e + (size_t)c * CB * D2 * HW, (ll)D2 * HW, HW,
            nullptr, 0, 0, D1);
    }

    // 3) xnT = normalize_columns(x1e)   (XT/H1T region now dead -> PPM reuse)
    colnorm_kernel<<<dim3(HW / 8, BATCH), blk, 0, stream>>>(out_x1e, xnT);

    // 4) S = relu(xn^T @ xn)^2          [xnT is both A and B]
    gemm_wmma<0, 2, false, false, true>
        <<<dim3(HW / BN, HW / BM, BATCH), blk, 0, stream>>>(
        xnT, (ll)HW * D2, D2, xnT, (ll)HW * D2, D2, nullptr,
        nullptr, 0, 0, Smat, (ll)HW * HW, HW, D2);

    // 5) g^T: [i,d] = sum_c x1eT[i,c] * Wp[d,c] + bp[d]; stored K-major -> g[d][i]
    gemm_wmma<2, 0, false, false, true>
        <<<dim3(D2 / BN, HW / BM, BATCH), blk, 0, stream>>>(
        x1eT, (ll)HW * D2, D2, wph, 0, D2, b_ppm,
        nullptr, 0, 0, g, (ll)D2 * HW, HW, D2);

    // 6) y^T: [i,c] = sum_j S[i,j] * g[c,j]; f32 K-major store -> y[b][c][i]
    gemm_wmma<0, 0, false, true, false>
        <<<dim3(D2 / BN, HW / BM, BATCH), blk, 0, stream>>>(
        Smat, (ll)HW * HW, HW, g, (ll)D2 * HW, HW, nullptr,
        out_y, (ll)D2 * HW, 0, nullptr, 0, HW, HW);

    // 7) A matrix
    amat_kernel<<<dim3(64, 64), dim3(16, 16), 0, stream>>>(p1, p2, out_A);
}
